// MultiHeadLatentAttention_87402584474001
// MI455X (gfx1250) — matrix operand order repaired and checked by reading, not compiled, not run on hardware
//
#include <hip/hip_runtime.h>

// ---------------------------------------------------------------------------
// MLA forward for gfx1250 (MI455X): all matmuls via v_wmma_f32_16x16x32_bf16.
// fp32 source data converted to bf16 on the fly (fused into LDS staging);
// conversions expressed as IR fptrunc so clang emits native bf16 cvt ops.
// K-tile staging in attention uses global_load_async_to_lds_b128 (ASYNCcnt).
// GEMM software-pipelines global loads (tile k+1) under WMMA compute (tile k).
// ---------------------------------------------------------------------------

#define HEADS 16
#define DH    128
#define RD    64
#define DTOT  192      // DH + RD
#define LSEQ  2048
#define EMB   2048
#define RK    512

typedef __attribute__((ext_vector_type(16))) __bf16 bf16x16;
typedef __attribute__((ext_vector_type(8)))  float  f32x8;
typedef __attribute__((ext_vector_type(2)))  float  f32x2v;
typedef __attribute__((ext_vector_type(2)))  __bf16 bf16x2;
typedef __attribute__((ext_vector_type(4)))  float  f32x4v;
typedef __attribute__((ext_vector_type(4)))  __bf16 bf16x4;

union BFrag {
    bf16x16 v;
    uint4   q[2];
};

// pack two fp32 -> packed bf16 pair (lo = x, hi = y); lowered as fptrunc
static __device__ __forceinline__ unsigned pack2bf(float x, float y) {
    f32x2v f;
    f.x = x; f.y = y;
    bf16x2 b = __builtin_convertvector(f, bf16x2);
    unsigned u;
    __builtin_memcpy(&u, &b, 4);
    return u;
}

// pack four fp32 -> four bf16 (element order preserved)
static __device__ __forceinline__ uint2 pack4bf(float4 v) {
    f32x4v f;
    f.x = v.x; f.y = v.y; f.z = v.z; f.w = v.w;
    bf16x4 b = __builtin_convertvector(f, bf16x4);
    uint2 u;
    __builtin_memcpy(&u, &b, 8);
    return u;
}

static __device__ __forceinline__ unsigned short f2bf(float f) {
    __bf16 b = (__bf16)f;
    unsigned short u;
    __builtin_memcpy(&u, &b, 2);
    return u;
}

// ---------------------------------------------------------------------------
// Generic C(f32) = A(f32,row-major) x B(f32,row-major) GEMM, bf16 WMMA inside.
// Workgroup tile 128x128, 8 waves (4 m x 2 n), wave tile 32x64, K-step 32.
// Pipeline: global float4 loads for next K-step issued before WMMA burst.
// ---------------------------------------------------------------------------
__global__ __launch_bounds__(256) void gemm_bf16_wmma(
    const float* __restrict__ A, int lda,
    const float* __restrict__ Bm, int ldb,
    float* __restrict__ C, int ldc,
    int M, int N, int K)
{
    __shared__ unsigned short sA[128 * 40];   // row-major m x k, padded
    __shared__ unsigned short sB[128 * 40];   // n-major:  n x k, padded

    const int t    = threadIdx.x;
    const int lane = t & 31;
    const int w    = t >> 5;
    const int hi16 = lane >> 4;     // 0 / 1
    const int l16  = lane & 15;
    const int wm   = w & 3;         // m offset wm*32
    const int wn   = w >> 2;        // n offset wn*64
    const int m0   = blockIdx.y * 128;
    const int n0   = blockIdx.x * 128;

    // A staging: thread covers rows rA+{0,32,64,96}, cols cA..cA+3 (float4)
    const int rA = t >> 3;            // 0..31
    const int cA = (t & 7) * 4;       // 0..28
    const float* aP = A + (size_t)(m0 + rA) * lda + cA;
    // B staging: thread covers k rows kB+{0,8,16,24}, cols nB..nB+3 (float4)
    const int kB = t >> 5;            // 0..7
    const int nB = (t & 31) * 4;      // 0..124
    const float* bP = Bm + (size_t)kB * ldb + n0 + nB;
    const bool bIn = (n0 + nB) < N;   // N always multiple of 4

    f32x8 acc[2][4];
#pragma unroll
    for (int mf = 0; mf < 2; ++mf)
#pragma unroll
        for (int nf = 0; nf < 4; ++nf)
#pragma unroll
            for (int j = 0; j < 8; ++j) acc[mf][nf][j] = 0.0f;

    // ---- pipeline prologue: fetch K-step 0 into registers ----
    float4 avr[4], bvr[4];
#pragma unroll
    for (int i = 0; i < 4; ++i)
        avr[i] = *(const float4*)(aP + (size_t)(32 * i) * lda);
#pragma unroll
    for (int i = 0; i < 4; ++i)
        bvr[i] = bIn ? *(const float4*)(bP + (size_t)(8 * i) * ldb)
                     : make_float4(0.f, 0.f, 0.f, 0.f);
    aP += 32;
    bP += (size_t)32 * ldb;

    for (int k0 = 0; k0 < K; k0 += 32) {
        // stage current tile registers -> LDS (convert to bf16)
#pragma unroll
        for (int i = 0; i < 4; ++i)
            *(uint2*)&sA[(rA + 32 * i) * 40 + cA] = pack4bf(avr[i]);
#pragma unroll
        for (int i = 0; i < 4; ++i) {
            const int kk = kB + 8 * i;
            sB[(nB + 0) * 40 + kk] = f2bf(bvr[i].x);
            sB[(nB + 1) * 40 + kk] = f2bf(bvr[i].y);
            sB[(nB + 2) * 40 + kk] = f2bf(bvr[i].z);
            sB[(nB + 3) * 40 + kk] = f2bf(bvr[i].w);
        }
        __syncthreads();

        // issue global loads for next K-step (in flight under the WMMAs)
        if (k0 + 32 < K) {
#pragma unroll
            for (int i = 0; i < 4; ++i)
                avr[i] = *(const float4*)(aP + (size_t)(32 * i) * lda);
#pragma unroll
            for (int i = 0; i < 4; ++i)
                bvr[i] = bIn ? *(const float4*)(bP + (size_t)(8 * i) * ldb)
                             : make_float4(0.f, 0.f, 0.f, 0.f);
            aP += 32;
            bP += (size_t)32 * ldb;
        }

        BFrag af[2], bf4[4];
#pragma unroll
        for (int mf = 0; mf < 2; ++mf) {
            const unsigned short* p = &sA[(wm * 32 + mf * 16 + l16) * 40 + (hi16 ? 8 : 0)];
            af[mf].q[0] = *(const uint4*)(p);
            af[mf].q[1] = *(const uint4*)(p + 16);
        }
#pragma unroll
        for (int nf = 0; nf < 4; ++nf) {
            const unsigned short* p = &sB[(wn * 64 + nf * 16 + l16) * 40 + (hi16 ? 8 : 0)];
            bf4[nf].q[0] = *(const uint4*)(p);
            bf4[nf].q[1] = *(const uint4*)(p + 16);
        }
#pragma unroll
        for (int nf = 0; nf < 4; ++nf)
#pragma unroll
            for (int mf = 0; mf < 2; ++mf)
                acc[mf][nf] = __builtin_amdgcn_wmma_f32_16x16x32_bf16(
                    false, af[mf].v, false, bf4[nf].v, (short)0, acc[mf][nf], false, false);
        __syncthreads();
    }

#pragma unroll
    for (int mf = 0; mf < 2; ++mf)
#pragma unroll
        for (int nf = 0; nf < 4; ++nf) {
            int col = n0 + wn * 64 + nf * 16 + l16;
            if (col < N) {
#pragma unroll
                for (int j = 0; j < 8; ++j) {
                    int row = m0 + wm * 32 + mf * 16 + j + (hi16 ? 8 : 0);
                    C[(size_t)row * ldc + col] = acc[mf][nf][j];
                }
            }
        }
}

// ---------------------------------------------------------------------------
// RoPE on q, fold in 1/sqrt(192), write Q as bf16 in (b,h,l,192) layout.
// ---------------------------------------------------------------------------
__global__ void rope_q_kernel(const float* __restrict__ q_all,
                              const float* __restrict__ cosT,
                              const float* __restrict__ sinT,
                              unsigned short* __restrict__ Qb, int total)
{
    int idx = blockIdx.x * 256 + threadIdx.x;
    if (idx >= total) return;
    int k   = idx % 160;
    int rem = idx / 160;
    int h   = rem % HEADS; rem /= HEADS;
    int l   = rem % LSEQ;
    int b   = rem / LSEQ;
    const float scale = 0.07216878364870323f;  // 1/sqrt(192)
    const float* src = q_all + ((size_t)(b * LSEQ + l) * HEADS + h) * DTOT;
    unsigned short* dst = Qb + ((size_t)(b * HEADS + h) * LSEQ + l) * DTOT;
    if (k < DH) {
        dst[k] = f2bf(src[k] * scale);
    } else {
        int p = k - DH;  // 0..31
        float c = cosT[l * 32 + p], s = sinT[l * 32 + p];
        float x1 = src[DH + 2 * p], x2 = src[DH + 2 * p + 1];
        *(unsigned*)(dst + DH + 2 * p) =
            pack2bf((x1 * c - x2 * s) * scale, (x1 * s + x2 * c) * scale);
    }
}

__global__ void rope_k_kernel(const float* __restrict__ kv_lat,
                              const float* __restrict__ cosT,
                              const float* __restrict__ sinT,
                              float* __restrict__ krope, int total)
{
    int idx = blockIdx.x * 256 + threadIdx.x;
    if (idx >= total) return;
    int p   = idx % 32;
    int row = idx / 32;                 // b*L + l
    int l   = row % LSEQ;
    float c = cosT[l * 32 + p], s = sinT[l * 32 + p];
    const float* src = kv_lat + (size_t)row * (RK + RD) + RK;
    float x1 = src[2 * p], x2 = src[2 * p + 1];
    krope[(size_t)row * RD + 2 * p]     = x1 * c - x2 * s;
    krope[(size_t)row * RD + 2 * p + 1] = x1 * s + x2 * c;
}

// K = [k_head | k_rope bcast], V = [v_head | v_rope] -> bf16 (b,h,l,192)
__global__ void build_kv_kernel(const float* __restrict__ up,
                                const float* __restrict__ krope,
                                unsigned short* __restrict__ Kb,
                                unsigned short* __restrict__ Vb, int total)
{
    int idx = blockIdx.x * 256 + threadIdx.x;
    if (idx >= total) return;
    int k   = idx % DTOT;
    int rem = idx / DTOT;
    int h   = rem % HEADS; rem /= HEADS;
    int l   = rem % LSEQ;
    int b   = rem / LSEQ;
    int row = b * LSEQ + l;
    const float* u = up + (size_t)row * 5120 + h * 320;
    size_t o = ((size_t)(b * HEADS + h) * LSEQ + l) * DTOT + k;
    float kv = (k < DH) ? u[k] : krope[(size_t)row * RD + (k - DH)];
    Kb[o] = f2bf(kv);
    Vb[o] = f2bf(u[DH + k]);          // covers v_head (128) then v_rope (64)
}

// ---------------------------------------------------------------------------
// Flash attention: workgroup = (b,h) x 128-query tile, 8 waves x 16 rows.
// K tile staged via async global->LDS DMA (ASYNCcnt); V staged transposed.
// S = Q K^T and O += P V both through bf16 WMMA; online softmax in registers.
// ---------------------------------------------------------------------------
__global__ __launch_bounds__(256) void mla_attn_kernel(
    const unsigned short* __restrict__ Qb,
    const unsigned short* __restrict__ Kb,
    const unsigned short* __restrict__ Vb,
    float* __restrict__ attn_out)
{
    __shared__ unsigned short sK[32 * 200];       // s-major (natural B operand)
    __shared__ unsigned short sV[192 * 40];       // V^T: d-major (B operand)
    __shared__ unsigned short sP[8 * 16 * 40];    // per-wave P scratch

    const int t     = threadIdx.x;
    const int lane  = t & 31;
    const int w     = t >> 5;
    const int hi16  = lane >> 4;
    const int l16   = lane & 15;
    const int bh    = blockIdx.y;                 // b*HEADS + h
    const int qtile = blockIdx.x;
    const int qbase = qtile * 128 + w * 16;

    const unsigned short* Qg = Qb + (size_t)bh * LSEQ * DTOT;
    const unsigned short* Kg = Kb + (size_t)bh * LSEQ * DTOT;
    const unsigned short* Vg = Vb + (size_t)bh * LSEQ * DTOT;

    // Q fragments: 6 chunks of k=32 across d=192, kept in VGPRs for all tiles
    BFrag qf[6];
    {
        const unsigned short* qrow = Qg + (size_t)(qbase + l16) * DTOT;
        int kh = hi16 ? 8 : 0;
#pragma unroll
        for (int c = 0; c < 6; ++c) {
            qf[c].q[0] = *(const uint4*)(qrow + c * 32 + kh);
            qf[c].q[1] = *(const uint4*)(qrow + c * 32 + 16 + kh);
        }
    }

    f32x8 o[12];
#pragma unroll
    for (int i = 0; i < 12; ++i)
#pragma unroll
        for (int j = 0; j < 8; ++j) o[i][j] = 0.0f;
    float mrow[8], lrow[8];
#pragma unroll
    for (int j = 0; j < 8; ++j) { mrow[j] = -3.0e38f; lrow[j] = 0.0f; }

    const int nst = qtile * 4 + 4;                // causal: s tiles 0..qtile*128+127
    for (int st = 0; st < nst; ++st) {
        const int s0 = st * 32;
        // ---- K tile: async DMA global -> LDS, 16B chunks (row pad handled) --
        {
            const unsigned short* ksrc = Kg + (size_t)s0 * DTOT;
#pragma unroll
            for (int i = 0; i < 2; ++i) {
                int idx = t + 256 * i;            // 32 rows x 12 chunks = 384
                if (idx < 384) {
                    int s = idx / 12, c = idx % 12;
                    unsigned long long ga =
                        (unsigned long long)(ksrc + (size_t)s * DTOT + c * 8);
                    unsigned ldsoff =
                        (unsigned)(unsigned long long)(const void*)&sK[s * 200 + c * 8];
                    asm volatile("global_load_async_to_lds_b128 %0, %1, off"
                                 :: "v"(ldsoff), "v"(ga) : "memory");
                }
            }
        }
        if (st + 1 < nst)                         // prefetch next V tile
            __builtin_prefetch(Vg + (size_t)(s0 + 32) * DTOT, 0, 1);
        // ---- V tile: vectorized read, transposed store (d-major) -----------
        {
            const unsigned short* vsrc = Vg + (size_t)s0 * DTOT;
#pragma unroll
            for (int i = 0; i < 3; ++i) {
                int idx = t + 256 * i;            // 32 rows x 24 oct-chunks
                int s = idx / 24, d8 = idx % 24;
                uint4 vv = *(const uint4*)(vsrc + (size_t)s * DTOT + d8 * 8);
                const unsigned short* pv = (const unsigned short*)&vv;
#pragma unroll
                for (int j = 0; j < 8; ++j)
                    sV[(d8 * 8 + j) * 40 + s] = pv[j];
            }
        }
        asm volatile("s_wait_asynccnt 0" ::: "memory");
        __syncthreads();

        if (s0 <= qbase + 15) {                   // tile not fully masked for this wave
            f32x8 S[2];
#pragma unroll
            for (int g = 0; g < 2; ++g)
#pragma unroll
                for (int j = 0; j < 8; ++j) S[g][j] = 0.0f;

#pragma unroll
            for (int c = 0; c < 6; ++c) {
#pragma unroll
                for (int g = 0; g < 2; ++g) {
                    BFrag kb;
                    const unsigned short* p = &sK[(g * 16 + l16) * 200 + c * 32 + (hi16 ? 8 : 0)];
                    kb.q[0] = *(const uint4*)(p);
                    kb.q[1] = *(const uint4*)(p + 16);
                    S[g] = __builtin_amdgcn_wmma_f32_16x16x32_bf16(
                        false, qf[c].v, false, kb.v, (short)0, S[g], false, false);
                }
            }
            // causal mask (softmax scale already folded into Q)
            if (s0 + 31 > qbase) {
#pragma unroll
                for (int g = 0; g < 2; ++g) {
                    int col = s0 + g * 16 + l16;
#pragma unroll
                    for (int j = 0; j < 8; ++j) {
                        int row = qbase + j + (hi16 ? 8 : 0);
                        if (col > row) S[g][j] += -1.0e9f;
                    }
                }
            }
            // online softmax: row stats via xor-shuffles within 16-lane halves
            float corr[8];
#pragma unroll
            for (int j = 0; j < 8; ++j) {
                float v = fmaxf(S[0][j], S[1][j]);
                v = fmaxf(v, __shfl_xor(v, 1, 32));
                v = fmaxf(v, __shfl_xor(v, 2, 32));
                v = fmaxf(v, __shfl_xor(v, 4, 32));
                v = fmaxf(v, __shfl_xor(v, 8, 32));
                float mnew = fmaxf(mrow[j], v);
                corr[j] = __expf(mrow[j] - mnew);
                mrow[j] = mnew;
                S[0][j] = __expf(S[0][j] - mnew);
                S[1][j] = __expf(S[1][j] - mnew);
                float rs = S[0][j] + S[1][j];
                rs += __shfl_xor(rs, 1, 32);
                rs += __shfl_xor(rs, 2, 32);
                rs += __shfl_xor(rs, 4, 32);
                rs += __shfl_xor(rs, 8, 32);
                lrow[j] = lrow[j] * corr[j] + rs;
            }
#pragma unroll
            for (int i = 0; i < 12; ++i)
#pragma unroll
                for (int j = 0; j < 8; ++j) o[i][j] *= corr[j];

            // P (f32 C-layout) -> bf16 A-layout via per-wave LDS scratch
            unsigned short* pw = &sP[w * 16 * 40];
#pragma unroll
            for (int g = 0; g < 2; ++g)
#pragma unroll
                for (int j = 0; j < 8; ++j) {
                    int row = j + (hi16 ? 8 : 0);
                    pw[row * 40 + g * 16 + l16] = f2bf(S[g][j]);
                }
            asm volatile("s_wait_dscnt 0" ::: "memory");
            BFrag pf;
            {
                const unsigned short* pr = pw + l16 * 40 + (hi16 ? 8 : 0);
                pf.q[0] = *(const uint4*)(pr);
                pf.q[1] = *(const uint4*)(pr + 16);
            }
            // O += P @ V
#pragma unroll
            for (int c2 = 0; c2 < 12; ++c2) {
                BFrag vb;
                const unsigned short* p = &sV[(c2 * 16 + l16) * 40 + (hi16 ? 8 : 0)];
                vb.q[0] = *(const uint4*)(p);
                vb.q[1] = *(const uint4*)(p + 16);
                o[c2] = __builtin_amdgcn_wmma_f32_16x16x32_bf16(
                    false, pf.v, false, vb.v, (short)0, o[c2], false, false);
            }
        }
        __syncthreads();
    }

    // normalize, write fp32 in (b,l,h,d) layout for the output GEMM
    const int b = bh / HEADS, h = bh % HEADS;
#pragma unroll
    for (int j = 0; j < 8; ++j) {
        int row = qbase + j + (hi16 ? 8 : 0);
        float inv = 1.0f / lrow[j];
        float* dst = attn_out + (((size_t)b * LSEQ + row) * HEADS + h) * DTOT;
#pragma unroll
        for (int c2 = 0; c2 < 12; ++c2)
            dst[c2 * 16 + l16] = o[c2][j] * inv;
    }
}

// ---------------------------------------------------------------------------
extern "C" void kernel_launch(void* const* d_in, const int* in_sizes, int n_in,
                              void* d_out, int out_size, void* d_ws, size_t ws_size,
                              hipStream_t stream)
{
    const float* x    = (const float*)d_in[0];
    const float* cosT = (const float*)d_in[1];
    const float* sinT = (const float*)d_in[2];
    const float* wq   = (const float*)d_in[3];
    const float* wkv  = (const float*)d_in[4];
    const float* wup  = (const float*)d_in[5];
    const float* wout = (const float*)d_in[6];
    float* out = (float*)d_out;

    const int BATCH = 2;
    const int BL = BATCH * LSEQ;                  // 4096

    // workspace layout (~210 MB); attn output aliases dead q_all buffer
    float* q_all  = (float*)d_ws;                               // BL x 3072
    float* kv_lat = q_all  + (size_t)BL * 3072;                 // BL x 576
    float* up     = kv_lat + (size_t)BL * (RK + RD);            // BL x 5120
    float* krope  = up     + (size_t)BL * 5120;                 // BL x 64
    unsigned short* Qb = (unsigned short*)(krope + (size_t)BL * RD);
    unsigned short* Kb = Qb + (size_t)BL * HEADS * DTOT;
    unsigned short* Vb = Kb + (size_t)BL * HEADS * DTOT;
    float* attn = q_all;                                        // alias (q_all dead)

    dim3 blk(256);

    // 1) q_all = x @ wq
    gemm_bf16_wmma<<<dim3(3072 / 128, BL / 128), blk, 0, stream>>>(
        x, EMB, wq, 3072, q_all, 3072, BL, 3072, EMB);
    // 2) kv_lat = x @ wkv_down
    gemm_bf16_wmma<<<dim3((RK + RD + 127) / 128, BL / 128), blk, 0, stream>>>(
        x, EMB, wkv, RK + RD, kv_lat, RK + RD, BL, RK + RD, EMB);
    // 3) RoPE q -> Q bf16 (b,h,l,192), scale folded in
    {
        int tot = BL * HEADS * 160;
        rope_q_kernel<<<(tot + 255) / 256, blk, 0, stream>>>(q_all, cosT, sinT, Qb, tot);
    }
    // 4) RoPE k_rope -> fp32 scratch
    {
        int tot = BL * 32;
        rope_k_kernel<<<(tot + 255) / 256, blk, 0, stream>>>(kv_lat, cosT, sinT, krope, tot);
    }
    // 5) up = c_kv @ w_up  (c_kv = kv_lat[:, :512] via lda stride, no copy)
    gemm_bf16_wmma<<<dim3(5120 / 128, BL / 128), blk, 0, stream>>>(
        kv_lat, RK + RD, wup, 5120, up, 5120, BL, 5120, RK);
    // 6) assemble K/V bf16 (b,h,l,192)
    {
        int tot = BL * HEADS * DTOT;
        build_kv_kernel<<<(tot + 255) / 256, blk, 0, stream>>>(up, krope, Kb, Vb, tot);
    }
    // 7) causal flash attention
    mla_attn_kernel<<<dim3(LSEQ / 128, BATCH * HEADS), blk, 0, stream>>>(Qb, Kb, Vb, attn);
    // 8) out = attn @ w_out
    gemm_bf16_wmma<<<dim3(EMB / 128, BL / 128), blk, 0, stream>>>(
        attn, 3072, wout, EMB, out, EMB, BL, EMB, 3072);
}
